// SpectralQuaternionAttention_55886114455827
// MI455X (gfx1250) — matrix operand order repaired and checked by reading, not compiled, not run on hardware
//
#include <hip/hip_runtime.h>
#include <hip/hip_bf16.h>
#include <math.h>

// ---------------------------------------------------------------------------
// SpectralQuaternionAttention on MI455X (gfx1250, wave32, WMMA)
//
// B=4 S=1024 E=768 H=12 D=64.  ~90 GFLOP total, ~35MB HBM traffic ->
// compute bound; all GEMM stages use v_wmma_f32_16x16x32_f16 with 32x64
// register tiles and ping-pong (move-free) software pipelining.
// __launch_bounds__(128,1) lifts the VGPR cap so the pipeline stays
// spill-free.
// spectral filter == circulant matmul with hr = Re(IFFT(exp(i*phase))).
// ---------------------------------------------------------------------------

#define S_LEN 1024
#define E_DIM 768
#define NB    4
#define NH    12
#define DH    64
#define BH    (NB*NH)         // 48
#define MROWS (NB*S_LEN)      // 4096
#define NCOLS (3*NB*E_DIM)    // 27648 circulant-GEMM columns
#define PI_F  3.14159265358979323846f

typedef __attribute__((ext_vector_type(16))) _Float16 v16h;
typedef __attribute__((ext_vector_type(8)))  float    v8f;

// ---------------- WMMA helpers (CDNA5 16x16x32 f16 -> f32) -----------------

__device__ __forceinline__ v8f wmma16(v16h a, v16h b, v8f c) {
  return __builtin_amdgcn_wmma_f32_16x16x32_f16(
      /*neg_a=*/false, a, /*neg_b=*/false, b,
      /*c_mod=*/(short)0, c, /*reuse_a=*/false, /*reuse_b=*/false);
}

// A fragment (16x32, MxK): lane m<16 holds row m with K {0..7,16..23},
// lane m+16 holds row m with K {8..15,24..31}.  base -> (row0,k0), ld halves.
__device__ __forceinline__ v16h load_a16(const _Float16* base, int ld) {
  const int lane = threadIdx.x & 31;
  const _Float16* p = base + (size_t)(lane & 15) * ld + ((lane >> 4) << 3);
  v16h r;
#pragma unroll
  for (int i = 0; i < 8; ++i) r[i] = p[i];
#pragma unroll
  for (int i = 0; i < 8; ++i) r[8 + i] = p[16 + i];
  return r;
}

// B fragment (32x16, KxN): lane n<16 holds column n, K {0..7,16..23};
// lane n+16 holds column n, K {8..15,24..31}.  Memory source: column n of B must be
// contiguous at base + n*ld (all our B sources are stored that way).
__device__ __forceinline__ v16h load_b16(const _Float16* base, int ld) {
  const int lane = threadIdx.x & 31;
  const _Float16* p = base + (size_t)(lane & 15) * ld + ((lane >> 4) << 3);
  v16h r;
#pragma unroll
  for (int i = 0; i < 8; ++i) r[i] = p[i];
#pragma unroll
  for (int i = 0; i < 8; ++i) r[8 + i] = p[16 + i];
  return r;
}

// 32x64 register-tiled GEMM core with ping-pong double buffering (no
// register-to-register fragment moves -> no WMMA WAR hazard nop bursts).
// A: 32 rows at A0 (lda), B: 64 columns at B0 (column n contiguous at n*ldb).
template <int KDIM>
__device__ __forceinline__ void gemm_core_32x64(const _Float16* __restrict__ A0, int lda,
                                                const _Float16* __restrict__ B0, int ldb,
                                                v8f acc[8]) {
  // buffer set 0 <- k-step 0
  v16h a0 = load_a16(A0, lda);
  v16h a1 = load_a16(A0 + (size_t)16 * lda, lda);
  v16h b0 = load_b16(B0, ldb);
  v16h b1 = load_b16(B0 + (size_t)16 * ldb, ldb);
  v16h b2 = load_b16(B0 + (size_t)32 * ldb, ldb);
  v16h b3 = load_b16(B0 + (size_t)48 * ldb, ldb);
  // steps are even (KDIM/32 = 24 or 32); loop consumes 2 steps/iteration
#pragma unroll 1
  for (int k = 32; k + 32 < KDIM; k += 64) {
    // buffer set 1 <- step k
    v16h c0 = load_a16(A0 + k, lda);
    v16h c1 = load_a16(A0 + (size_t)16 * lda + k, lda);
    v16h d0 = load_b16(B0 + k, ldb);
    v16h d1 = load_b16(B0 + (size_t)16 * ldb + k, ldb);
    v16h d2 = load_b16(B0 + (size_t)32 * ldb + k, ldb);
    v16h d3 = load_b16(B0 + (size_t)48 * ldb + k, ldb);
    acc[0] = wmma16(a0, b0, acc[0]);
    acc[1] = wmma16(a0, b1, acc[1]);
    acc[2] = wmma16(a0, b2, acc[2]);
    acc[3] = wmma16(a0, b3, acc[3]);
    acc[4] = wmma16(a1, b0, acc[4]);
    acc[5] = wmma16(a1, b1, acc[5]);
    acc[6] = wmma16(a1, b2, acc[6]);
    acc[7] = wmma16(a1, b3, acc[7]);
    // buffer set 0 <- step k+32
    a0 = load_a16(A0 + k + 32, lda);
    a1 = load_a16(A0 + (size_t)16 * lda + k + 32, lda);
    b0 = load_b16(B0 + k + 32, ldb);
    b1 = load_b16(B0 + (size_t)16 * ldb + k + 32, ldb);
    b2 = load_b16(B0 + (size_t)32 * ldb + k + 32, ldb);
    b3 = load_b16(B0 + (size_t)48 * ldb + k + 32, ldb);
    acc[0] = wmma16(c0, d0, acc[0]);
    acc[1] = wmma16(c0, d1, acc[1]);
    acc[2] = wmma16(c0, d2, acc[2]);
    acc[3] = wmma16(c0, d3, acc[3]);
    acc[4] = wmma16(c1, d0, acc[4]);
    acc[5] = wmma16(c1, d1, acc[5]);
    acc[6] = wmma16(c1, d2, acc[6]);
    acc[7] = wmma16(c1, d3, acc[7]);
  }
  // epilogue: last step (KDIM-32) in buffer set 1, then drain both
  {
    constexpr int k = KDIM - 32;
    v16h c0 = load_a16(A0 + k, lda);
    v16h c1 = load_a16(A0 + (size_t)16 * lda + k, lda);
    v16h d0 = load_b16(B0 + k, ldb);
    v16h d1 = load_b16(B0 + (size_t)16 * ldb + k, ldb);
    v16h d2 = load_b16(B0 + (size_t)32 * ldb + k, ldb);
    v16h d3 = load_b16(B0 + (size_t)48 * ldb + k, ldb);
    acc[0] = wmma16(a0, b0, acc[0]);
    acc[1] = wmma16(a0, b1, acc[1]);
    acc[2] = wmma16(a0, b2, acc[2]);
    acc[3] = wmma16(a0, b3, acc[3]);
    acc[4] = wmma16(a1, b0, acc[4]);
    acc[5] = wmma16(a1, b1, acc[5]);
    acc[6] = wmma16(a1, b2, acc[6]);
    acc[7] = wmma16(a1, b3, acc[7]);
    acc[0] = wmma16(c0, d0, acc[0]);
    acc[1] = wmma16(c0, d1, acc[1]);
    acc[2] = wmma16(c0, d2, acc[2]);
    acc[3] = wmma16(c0, d3, acc[3]);
    acc[4] = wmma16(c1, d0, acc[4]);
    acc[5] = wmma16(c1, d1, acc[5]);
    acc[6] = wmma16(c1, d2, acc[6]);
    acc[7] = wmma16(c1, d3, acc[7]);
  }
}

// ---------------- stage 1: spectral kernel hr (1024 taps) ------------------

__global__ void k_hr(float* __restrict__ hr) {
  int m = blockIdx.x * blockDim.x + threadIdx.x;
  if (m >= S_LEN) return;
  float acc = 0.f;
  for (int k = 0; k < S_LEN; ++k) {
    float freq = (float)(k < S_LEN / 2 ? k : k - S_LEN) / (float)S_LEN;
    float kk = 2.0f * PI_F * freq;
    float sg = (kk > 0.f) ? 1.f : ((kk < 0.f) ? -1.f : 0.f);
    float ph = sg * atanf(logf(fabsf(kk) + 1e-10f));
    int km = (k * m) & (S_LEN - 1);                 // exact angle reduction
    acc += cosf((2.0f * PI_F / (float)S_LEN) * (float)km + ph);
  }
  hr[m] = acc * (1.0f / (float)S_LEN);
}

__global__ void k_buildH(const float* __restrict__ hr, _Float16* __restrict__ Hm) {
  int i = blockIdx.x * blockDim.x + threadIdx.x;
  int s = i >> 10, t = i & (S_LEN - 1);
  Hm[i] = (_Float16)hr[(s - t) & (S_LEN - 1)];
}

// ---------------- stage 2: f32 -> f16 casts --------------------------------

__global__ void k_cast(_Float16* __restrict__ dst, const float* __restrict__ src, int n) {
  int i = blockIdx.x * blockDim.x + threadIdx.x;
  if (i < n) dst[i] = (_Float16)src[i];
}

// ---------------- stage 3: QKV projection GEMM -----------------------------
// q/k/v[m][n] = sum_k Xh[m][k] * W[n][k] + bias[n]   (W row n == B column n)
// one wave -> 32x64 output tile

__global__ void __launch_bounds__(128, 1)
k_qkv_gemm(const _Float16* __restrict__ Xh,
           const _Float16* __restrict__ Wh,   // [3][768][768]
           const float* __restrict__ bq,
           const float* __restrict__ bk,
           const float* __restrict__ bv,
           float* __restrict__ qkv) {         // [3][4096][768]
  const int wv  = (blockIdx.x * blockDim.x + threadIdx.x) >> 5;
  const int mat = wv / (128 * 12);
  const int rem = wv % (128 * 12);
  const int m0  = (rem % 128) * 32;
  const int n0  = (rem / 128) * 64;
  const _Float16* A0 = Xh + (size_t)m0 * E_DIM;
  const _Float16* B0 = Wh + (size_t)mat * E_DIM * E_DIM + (size_t)n0 * E_DIM;
  v8f acc[8] = {};
  gemm_core_32x64<E_DIM>(A0, E_DIM, B0, E_DIM, acc);

  const float* bias = (mat == 0) ? bq : (mat == 1) ? bk : bv;
  float* out = qkv + (size_t)mat * MROWS * E_DIM;
  const int lane = threadIdx.x & 31;
  const int nl = lane & 15, mhi = (lane >> 4) << 3;
#pragma unroll
  for (int i = 0; i < 2; ++i)
#pragma unroll
    for (int j = 0; j < 4; ++j) {
      int n = n0 + j * 16 + nl;
      float bb = bias[n];
#pragma unroll
      for (int g = 0; g < 8; ++g)
        out[(size_t)(m0 + i * 16 + mhi + g) * E_DIM + n] = acc[i * 4 + j][g] + bb;
    }
}

// ---------------- stage 4: per-feature sequence norms ----------------------

__global__ void k_colnorm(const float* __restrict__ qkv, float* __restrict__ invn) {
  int col = blockIdx.x * blockDim.x + threadIdx.x;     // 3*4*768 = 9216
  int mat = col / (NB * E_DIM);
  int r2  = col % (NB * E_DIM);
  int b   = r2 / E_DIM, e = r2 % E_DIM;
  const float* x = qkv + (size_t)mat * MROWS * E_DIM + (size_t)b * S_LEN * E_DIM + e;
  float sum = 0.f;
  for (int s = 0; s < S_LEN; ++s) { float v = x[(size_t)s * E_DIM]; sum += v * v; }
  invn[col] = 1.0f / sqrtf(sum + 1e-8f);
}

// ---------------- stage 5: quaternion combine -> Ct (transposed, f16) ------
// Ct[col][t],  col = mat*3072 + b*768 + e

__global__ void k_makect(const float* __restrict__ qkv, const float* __restrict__ invn,
                         const float* __restrict__ Wqr, _Float16* __restrict__ Ct) {
  const int col = blockIdx.x >> 2;
  const int t   = (blockIdx.x & 3) * 256 + threadIdx.x;
  const int mat = col / (NB * E_DIM);
  const int r2  = col % (NB * E_DIM);
  const int b   = r2 / E_DIM, e = r2 % E_DIM;
  const float* x = qkv + (size_t)mat * MROWS * E_DIM + (size_t)b * S_LEN * E_DIM + e;
  const float in = invn[col];
  const int tm = (t + S_LEN - 1) & (S_LEN - 1);
  const int tp = (t + 1) & (S_LEN - 1);
  float f0 = x[(size_t)t  * E_DIM] * in;
  float fm = x[(size_t)tm * E_DIM] * in;      // psi1 = 0.5*fn[t-1]
  float fp = x[(size_t)tp * E_DIM] * in;      // psi2 = 0.5*fn[t+1]
  float qn = sqrtf(f0 * f0 + 0.25f * fm * fm + 0.25f * fp * fp + 1e-8f);
  float c  = (Wqr[0] * f0 + 0.5f * Wqr[1] * fm + 0.5f * Wqr[2] * fp) / qn;
  Ct[(size_t)col * S_LEN + t] = (_Float16)c;
}

// ---------------- stage 6: circulant GEMM  r = H @ C + bqr -----------------
// writes qr/kr row-major [bh][s][64] (f16) and vr transposed [bh][64][s] (f16)
// one wave -> 32x64 output tile

__global__ void __launch_bounds__(128, 1)
k_circ_gemm(const _Float16* __restrict__ Hm,   // [1024][1024]
            const _Float16* __restrict__ Ct,   // [27648][1024]
            const float* __restrict__ bqr,
            _Float16* __restrict__ QR,
            _Float16* __restrict__ KR,
            _Float16* __restrict__ VRT) {
  const int wv = (blockIdx.x * blockDim.x + threadIdx.x) >> 5;
  const int s0 = (wv & 31) * 32;
  const int c0 = (wv >> 5) * 64;
  const _Float16* A0 = Hm + (size_t)s0 * S_LEN;
  const _Float16* B0 = Ct + (size_t)c0 * S_LEN;
  v8f acc[8] = {};
  gemm_core_32x64<S_LEN>(A0, S_LEN, B0, S_LEN, acc);

  const float bb = bqr[0];
  const int lane = threadIdx.x & 31;
  const int nl = lane & 15, mhi = (lane >> 4) << 3;
#pragma unroll
  for (int j = 0; j < 4; ++j) {
    int col = c0 + j * 16 + nl;
    int mat = col / (NB * E_DIM);
    int r2  = col % (NB * E_DIM);
    int b   = r2 / E_DIM, e = r2 % E_DIM;
    int h = e >> 6, d = e & 63;
    int bh = b * NH + h;
#pragma unroll
    for (int i = 0; i < 2; ++i)
#pragma unroll
      for (int g = 0; g < 8; ++g) {
        int s = s0 + i * 16 + mhi + g;
        _Float16 v = (_Float16)(acc[i * 4 + j][g] + bb);
        if (mat == 0)      QR [((size_t)bh * S_LEN + s) * DH + d] = v;
        else if (mat == 1) KR [((size_t)bh * S_LEN + s) * DH + d] = v;
        else               VRT[((size_t)bh * DH + d) * S_LEN + s] = v;
      }
  }
}

// ---------------- stage 7: flash attention (1 wave = 16 query rows) --------

#define PST 40   // padded LDS row stride (halves)

__global__ void __launch_bounds__(128, 1)
k_attn(const _Float16* __restrict__ QR,
       const _Float16* __restrict__ KR,
       const _Float16* __restrict__ VRT,
       _Float16* __restrict__ CTX) {            // [b][s][768] f16
  __shared__ _Float16 lp[4][16 * PST];
  const int wv = threadIdx.x >> 5;
  const int gw = blockIdx.x * 4 + wv;
  const int bh = gw >> 6;
  const int s0 = (gw & 63) * 16;
  const int lane = threadIdx.x & 31;
  const int nl = lane & 15, hi = lane >> 4, mhi = hi << 3;
  _Float16* myp = lp[wv];

  const _Float16* qb = QR + ((size_t)bh * S_LEN + s0) * DH;
  v16h aq0 = load_a16(qb, DH);        // d 0..31
  v16h aq1 = load_a16(qb + 32, DH);   // d 32..63

  float mrow[8], lrow[8];
#pragma unroll
  for (int g = 0; g < 8; ++g) { mrow[g] = -1e30f; lrow[g] = 0.f; }
  v8f accC[4] = {};

#pragma unroll 1
  for (int t0 = 0; t0 < S_LEN; t0 += 32) {
    const _Float16* kb = KR + ((size_t)bh * S_LEN + t0) * DH;
    v8f sc0 = {}, sc1 = {};
    { v16h b0 = load_b16(kb, DH);
      v16h b1 = load_b16(kb + 32, DH);
      v16h b2 = load_b16(kb + 16 * DH, DH);
      v16h b3 = load_b16(kb + 16 * DH + 32, DH);
      sc0 = wmma16(aq0, b0, sc0);
      sc0 = wmma16(aq1, b1, sc0);
      sc1 = wmma16(aq0, b2, sc1);
      sc1 = wmma16(aq1, b3, sc1); }

    float p0[8], p1[8], scl[8];
#pragma unroll
    for (int g = 0; g < 8; ++g) {
      float v0 = sc0[g] * 0.125f, v1 = sc1[g] * 0.125f;   // 1/sqrt(64)
      float mx = fmaxf(v0, v1);
#pragma unroll
      for (int off = 1; off < 16; off <<= 1)
        mx = fmaxf(mx, __shfl_xor(mx, off, 32));
      float mnew = fmaxf(mrow[g], mx);
      scl[g] = expf(mrow[g] - mnew);
      float e0 = expf(v0 - mnew), e1 = expf(v1 - mnew);
      float rs = e0 + e1;
#pragma unroll
      for (int off = 1; off < 16; off <<= 1)
        rs += __shfl_xor(rs, off, 32);
      lrow[g] = lrow[g] * scl[g] + rs;
      mrow[g] = mnew;
      p0[g] = e0; p1[g] = e1;
    }
#pragma unroll
    for (int j = 0; j < 4; ++j)
#pragma unroll
      for (int g = 0; g < 8; ++g) accC[j][g] *= scl[g];

    // stage P (16x32) through LDS to convert D-frag layout -> A-frag layout
#pragma unroll
    for (int g = 0; g < 8; ++g) {
      int r = mhi + g;
      myp[r * PST + nl]      = (_Float16)p0[g];
      myp[r * PST + 16 + nl] = (_Float16)p1[g];
    }
    asm volatile("s_wait_dscnt 0x0" ::: "memory");  // DS in-order per wave; belt+braces
    v16h pa = load_a16(myp, PST);

    const _Float16* vb = VRT + (size_t)bh * DH * S_LEN + t0;
#pragma unroll
    for (int j = 0; j < 4; ++j) {
      v16h bv = load_b16(vb + (size_t)(j * 16) * S_LEN, S_LEN);
      accC[j] = wmma16(pa, bv, accC[j]);
    }
  }

  float inv[8];
#pragma unroll
  for (int g = 0; g < 8; ++g) inv[g] = 1.0f / lrow[g];
  const int b = bh / NH, h = bh % NH;
#pragma unroll
  for (int j = 0; j < 4; ++j)
#pragma unroll
    for (int g = 0; g < 8; ++g) {
      int s = s0 + mhi + g;
      int e = h * DH + j * 16 + nl;
      CTX[((size_t)b * S_LEN + s) * E_DIM + e] = (_Float16)(accC[j][g] * inv[g]);
    }
}

// ---------------- stage 8: output projection -------------------------------
// one wave -> 32x64 output tile

__global__ void __launch_bounds__(128, 1)
k_out_gemm(const _Float16* __restrict__ CTX,
           const _Float16* __restrict__ Woh,   // [768][768]
           const float* __restrict__ bo,
           float* __restrict__ out) {          // [4096][768] f32
  const int wv = (blockIdx.x * blockDim.x + threadIdx.x) >> 5;
  const int m0 = (wv % 128) * 32;
  const int n0 = (wv / 128) * 64;
  const _Float16* A0 = CTX + (size_t)m0 * E_DIM;
  const _Float16* B0 = Woh + (size_t)n0 * E_DIM;
  v8f acc[8] = {};
  gemm_core_32x64<E_DIM>(A0, E_DIM, B0, E_DIM, acc);

  const int lane = threadIdx.x & 31;
  const int nl = lane & 15, mhi = (lane >> 4) << 3;
#pragma unroll
  for (int i = 0; i < 2; ++i)
#pragma unroll
    for (int j = 0; j < 4; ++j) {
      int n = n0 + j * 16 + nl;
      float bb = bo[n];
#pragma unroll
      for (int g = 0; g < 8; ++g)
        out[(size_t)(m0 + i * 16 + mhi + g) * E_DIM + n] = acc[i * 4 + j][g] + bb;
    }
}

// ---------------- host orchestration ---------------------------------------

extern "C" void kernel_launch(void* const* d_in, const int* in_sizes, int n_in,
                              void* d_out, int out_size, void* d_ws, size_t ws_size,
                              hipStream_t stream) {
  const float* hidden = (const float*)d_in[0];
  const float* Wq  = (const float*)d_in[1];
  const float* bq  = (const float*)d_in[2];
  const float* Wk  = (const float*)d_in[3];
  const float* bk  = (const float*)d_in[4];
  const float* Wv  = (const float*)d_in[5];
  const float* bv  = (const float*)d_in[6];
  const float* Wo  = (const float*)d_in[7];
  const float* bo  = (const float*)d_in[8];
  const float* Wqr = (const float*)d_in[9];
  const float* bqr = (const float*)d_in[10];
  float* out = (float*)d_out;

  char* w = (char*)d_ws;
  constexpr size_t SZ_HR  = 4096;                              // 1024 f32
  constexpr size_t SZ_HM  = (size_t)S_LEN * S_LEN * 2;         // 2 MB f16
  constexpr size_t SZ_XH  = (size_t)MROWS * E_DIM * 2;         // 6.3 MB
  constexpr size_t SZ_WH  = (size_t)4 * E_DIM * E_DIM * 2;     // 4.7 MB
  constexpr size_t SZ_CN  = (size_t)3 * NB * E_DIM * 4;        // 36 KB
  constexpr size_t SZ_CT  = (size_t)NCOLS * S_LEN * 2;         // 56.6 MB
  constexpr size_t SZ_HEAD = (size_t)BH * S_LEN * DH * 2;      // 6.3 MB each

  float*    HR  = (float*)w;
  _Float16* HM  = (_Float16*)(w + SZ_HR);
  _Float16* XH  = (_Float16*)(w + SZ_HR + SZ_HM);
  _Float16* WH  = (_Float16*)(w + SZ_HR + SZ_HM + SZ_XH);
  float*    CN  = (float*)(w + SZ_HR + SZ_HM + SZ_XH + SZ_WH);
  _Float16* CT  = (_Float16*)(w + SZ_HR + SZ_HM + SZ_XH + SZ_WH + SZ_CN);
  char*     qkvBase = w + SZ_HR + SZ_HM + SZ_XH + SZ_WH + SZ_CN + SZ_CT;
  float*    QKV = (float*)qkvBase;                             // 37.7 MB (dead after k_makect)
  _Float16* QR  = (_Float16*)qkvBase;                          // aliases into QKV region
  _Float16* KR  = (_Float16*)(qkvBase + SZ_HEAD);
  _Float16* VRT = (_Float16*)(qkvBase + 2 * SZ_HEAD);
  _Float16* CTX = (_Float16*)(qkvBase + 3 * SZ_HEAD);

  const int WE = E_DIM * E_DIM;                                // 589824

  // casts to f16
  k_cast<<<(MROWS * E_DIM + 255) / 256, 256, 0, stream>>>(XH, hidden, MROWS * E_DIM);
  k_cast<<<(WE + 255) / 256, 256, 0, stream>>>(WH + 0 * (size_t)WE, Wq, WE);
  k_cast<<<(WE + 255) / 256, 256, 0, stream>>>(WH + 1 * (size_t)WE, Wk, WE);
  k_cast<<<(WE + 255) / 256, 256, 0, stream>>>(WH + 2 * (size_t)WE, Wv, WE);
  k_cast<<<(WE + 255) / 256, 256, 0, stream>>>(WH + 3 * (size_t)WE, Wo, WE);

  // spectral circulant kernel
  k_hr<<<4, 256, 0, stream>>>(HR);
  k_buildH<<<(S_LEN * S_LEN) / 256, 256, 0, stream>>>(HR, HM);

  // QKV projections: 3*128*12 = 4608 waves (32x64 tiles), 4 waves/block
  k_qkv_gemm<<<1152, 128, 0, stream>>>(XH, WH, bq, bk, bv, QKV);

  // sequence norms + quaternion combine (transposed f16 output)
  k_colnorm<<<(3 * NB * E_DIM) / 256, 256, 0, stream>>>(QKV, CN);
  k_makect<<<NCOLS * 4, 256, 0, stream>>>(QKV, CN, Wqr, CT);

  // circulant GEMM: 32*432 = 13824 waves (32x64 tiles)
  k_circ_gemm<<<3456, 128, 0, stream>>>(HM, CT, bqr, QR, KR, VRT);

  // flash attention: 48*64 = 3072 waves
  k_attn<<<768, 128, 0, stream>>>(QR, KR, VRT, CTX);

  // output projection: 128*12 = 1536 waves (32x64 tiles)
  k_out_gemm<<<384, 128, 0, stream>>>(CTX, WH + 3 * (size_t)WE, bo, out);
}
